// ScalableDRN_59682865545745
// MI455X (gfx1250) — matrix operand order repaired and checked by reading, compiled, not run on hardware
//
#include <hip/hip_runtime.h>

// ---------------------------------------------------------------------------
// ScalableDRN on MI455X (gfx1250): bf16 WMMA (f32 accumulate) pipeline.
//   B=8192, IN=2048, C=1000, P1=8,N1=128, P2=4,N2=64
// ---------------------------------------------------------------------------

typedef __attribute__((ext_vector_type(16))) __bf16 v16bf;
typedef __attribute__((ext_vector_type(8)))  float  v8f;

#define BATCH 8192

__device__ __forceinline__ unsigned bf_bits(float f) {
  union { float f; unsigned u; } in; in.f = f;
  return (in.u + 0x7FFFu + ((in.u >> 16) & 1u)) >> 16;   // rne
}
__device__ __forceinline__ __bf16 f2bf(float f) {
  union { unsigned short s; __bf16 b; } out;
  out.s = (unsigned short)bf_bits(f);
  return out.b;
}
// pack two floats -> two bf16 in one dword (lo = a, hi = b)
__device__ __forceinline__ unsigned pack_bf2(float a, float b) {
  union { float f; unsigned u; } ub; ub.f = b;
  unsigned hb = (ub.u + 0x7FFFu + ((ub.u >> 16) & 1u)) & 0xFFFF0000u;
  return hb | bf_bits(a);
}

__device__ __forceinline__ float fast_tanh(float x) {
  return 1.f - 2.f / (1.f + __expf(2.f * x));
}

// A operand 16x32 bf16: lane&15 = M row; lanes 0-15 K chunks {0-7,16-23},
// lanes 16-31 K chunks {8-15,24-31}  (ISA 7.12.2, 16-bit A layout)
__device__ __forceinline__ v16bf load_fragA(const __bf16* base, int ld) {
  const int lane = threadIdx.x & 31;
  const __bf16* p = base + (lane & 15) * ld + ((lane >> 4) << 3);
  union { v16bf v; uint4 q[2]; } f;
  f.q[0] = *(const uint4*)(p);
  f.q[1] = *(const uint4*)(p + 16);
  return f.v;
}

// B operand 32x16 bf16: lane&15 = N col; lanes 0-15 K chunks {0-7,16-23},
// lanes 16-31 K chunks {8-15,24-31}  (same K order as A)
__device__ __forceinline__ v16bf load_fragB(const __bf16* base, int ld) {
  const int lane = threadIdx.x & 31;
  const __bf16* p = base + (lane & 15) * ld + ((lane >> 4) << 3);
  union { v16bf v; uint4 q[2]; } f;
  f.q[0] = *(const uint4*)(p);
  f.q[1] = *(const uint4*)(p + 16);
  return f.v;
}

__device__ __forceinline__ v8f wmma_bf16(v16bf a, v16bf b, v8f c) {
  return __builtin_amdgcn_wmma_f32_16x16x32_bf16(false, a, false, b,
                                                 (short)0, c, false, false);
}

// ---------------------------------------------------------------------------
// sel = softmax(X @ SW + SB)  ; one wave per row
// ---------------------------------------------------------------------------
template <int K, int P>
__global__ __launch_bounds__(256) void sel_kernel(
    const float* __restrict__ X, const float* __restrict__ SW,
    const float* __restrict__ SB, float* __restrict__ sel) {
  const int lane = threadIdx.x & 31;
  const int wave = threadIdx.x >> 5;
  const int row  = blockIdx.x * 8 + wave;
  float acc[P];
#pragma unroll
  for (int p = 0; p < P; ++p) acc[p] = 0.f;
  const float* xr = X + (size_t)row * K;
  for (int k = lane; k < K; k += 32) {
    float xv = xr[k];
#pragma unroll
    for (int c = 0; c < P / 4; ++c) {
      float4 w = *(const float4*)(SW + (size_t)k * P + c * 4);
      acc[c * 4 + 0] += xv * w.x;
      acc[c * 4 + 1] += xv * w.y;
      acc[c * 4 + 2] += xv * w.z;
      acc[c * 4 + 3] += xv * w.w;
    }
  }
#pragma unroll
  for (int p = 0; p < P; ++p) {
#pragma unroll
    for (int off = 16; off > 0; off >>= 1)
      acc[p] += __shfl_xor(acc[p], off, 32);
  }
  float mx = -3.0e38f;
#pragma unroll
  for (int p = 0; p < P; ++p) { float l = acc[p] + SB[p]; mx = l > mx ? l : mx; }
  float den = 0.f;
#pragma unroll
  for (int p = 0; p < P; ++p) den += __expf(acc[p] + SB[p] - mx);
  if (lane < P)
    sel[(size_t)row * P + lane] = __expf(acc[lane] + SB[lane] - mx) / den;
}

// ---------------------------------------------------------------------------
// One DRN layer, one partition per blockIdx.y, 128 rows per blockIdx.x.
// ---------------------------------------------------------------------------
template <int K, int P, int N>
__global__ __launch_bounds__(256) void drn_layer_kernel(
    const float* __restrict__ X, const float* __restrict__ W,
    const float* __restrict__ bias, const float* __restrict__ R,
    const float* __restrict__ I, const float* __restrict__ Ag,
    const float* __restrict__ sel, float* __restrict__ out) {
  constexpr int MT = 128, KT = 32, PAD = 8;
  constexpr int NT  = N / 16;
  constexpr int RKC = 64;
  constexpr int LDA = KT + PAD;        // 40
  constexpr int LDH = N + PAD;         // 136 / 72
  constexpr int LDR = RKC + PAD;       // 72
  constexpr int SM1 = MT * LDA + N * LDA;
  constexpr int SM2 = MT * LDH + N * LDR;
  constexpr int SM  = SM1 > SM2 ? SM1 : SM2;

  __shared__ __align__(16) __bf16 smem[SM];
  __shared__ float sel_s[MT];
  __bf16* lds_a = smem;                 // [MT][LDA]
  __bf16* lds_b = smem + MT * LDA;      // [N][LDA]   (transposed: [n][k])
  __bf16* lds_h = smem;                 // [MT][LDH]
  __bf16* lds_r = smem + MT * LDH;      // [N][LDR]   (transposed: [m_out][k])

  const int tid  = threadIdx.x;
  const int wave = tid >> 5;
  const int lane = tid & 31;
  const int nl   = lane & 15;
  const int mloc = (lane >> 4) * 8;
  const int p    = blockIdx.y;
  const int m0   = blockIdx.x * MT;
  const float* Wp = W + (size_t)p * K * N;
  const float* Rp = R + (size_t)p * N * N;
  const float* Ip = I + (size_t)p * N * N;

  if (tid < MT) sel_s[tid] = sel[(size_t)(m0 + tid) * P + p];

  const v8f vzero = {0.f, 0.f, 0.f, 0.f, 0.f, 0.f, 0.f, 0.f};
  v8f acc[NT];
#pragma unroll
  for (int n = 0; n < NT; ++n) acc[n] = vzero;

  // ---------------- main GEMM: X[MTxK] @ Wp[KxN] ----------------
  for (int kb = 0; kb < K; kb += KT) {
    __syncthreads();
    // A tile: float4 loads, packed b64 LDS stores  (MT*KT/4 = 1024 groups)
#pragma unroll
    for (int i = 0; i < (MT * KT / 4) / 256; ++i) {
      int idx = tid + i * 256;
      int m = idx >> 3, k4 = (idx & 7) << 2;
      float4 v = *(const float4*)(X + (size_t)(m0 + m) * K + kb + k4);
      uint2 pk; pk.x = pack_bf2(v.x, v.y); pk.y = pack_bf2(v.z, v.w);
      *(uint2*)(lds_a + m * LDA + k4) = pk;
    }
    // B tile (transpose): float4 loads, 4 scalar bf16 stores
#pragma unroll
    for (int i = 0; i < (KT * N / 4) / 256; ++i) {
      int idx = tid + i * 256;
      int k = idx / (N / 4), n4 = (idx % (N / 4)) << 2;
      float4 v = *(const float4*)(Wp + (size_t)(kb + k) * N + n4);
      lds_b[(n4 + 0) * LDA + k] = f2bf(v.x);
      lds_b[(n4 + 1) * LDA + k] = f2bf(v.y);
      lds_b[(n4 + 2) * LDA + k] = f2bf(v.z);
      lds_b[(n4 + 3) * LDA + k] = f2bf(v.w);
    }
    __syncthreads();
    v16bf af = load_fragA(lds_a + wave * 16 * LDA, LDA);
    v16bf bfr[NT];
#pragma unroll
    for (int n = 0; n < NT; ++n) bfr[n] = load_fragB(lds_b + n * 16 * LDA, LDA);
#pragma unroll
    for (int n = 0; n < NT; ++n) acc[n] = wmma_bf16(af, bfr[n], acc[n]);
  }
  __syncthreads();   // lds_a/lds_b dead; smem reused as lds_h/lds_r

  // ---------------- bias + relu, stash bf16 h in LDS ----------------
#pragma unroll
  for (int n = 0; n < NT; ++n) {
    float bv = bias[p * N + n * 16 + nl];
#pragma unroll
    for (int r = 0; r < 8; ++r) {
      float h = acc[n][r] + bv;
      h = h > 0.f ? h : 0.f;
      acc[n][r] = h;
      lds_h[(wave * 16 + mloc + r) * LDH + n * 16 + nl] = f2bf(h);
    }
  }

  // ---------------- t = h @ R[p] ; h = h + tanh(t) ----------------
  v8f acc2[NT];
#pragma unroll
  for (int n = 0; n < NT; ++n) acc2[n] = vzero;
  for (int kc = 0; kc < N; kc += RKC) {
    __syncthreads();
#pragma unroll
    for (int i = 0; i < (RKC * N / 4) / 256; ++i) {
      int idx = tid + i * 256;
      int k = idx / (N / 4), c4 = (idx % (N / 4)) << 2;
      float4 v = *(const float4*)(Rp + (size_t)(kc + k) * N + c4);
      lds_r[(c4 + 0) * LDR + k] = f2bf(v.x);
      lds_r[(c4 + 1) * LDR + k] = f2bf(v.y);
      lds_r[(c4 + 2) * LDR + k] = f2bf(v.z);
      lds_r[(c4 + 3) * LDR + k] = f2bf(v.w);
    }
    __syncthreads();
#pragma unroll
    for (int kb = 0; kb < RKC; kb += KT) {
      v16bf af = load_fragA(lds_h + wave * 16 * LDH + kc + kb, LDH);
      v16bf bfr[NT];
#pragma unroll
      for (int n = 0; n < NT; ++n)
        bfr[n] = load_fragB(lds_r + n * 16 * LDR + kb, LDR);
#pragma unroll
      for (int n = 0; n < NT; ++n) acc2[n] = wmma_bf16(af, bfr[n], acc2[n]);
    }
  }
#pragma unroll
  for (int n = 0; n < NT; ++n) {
#pragma unroll
    for (int r = 0; r < 8; ++r) {
      float h2 = acc[n][r] + fast_tanh(acc2[n][r]);
      acc[n][r] = h2;
      lds_h[(wave * 16 + mloc + r) * LDH + n * 16 + nl] = f2bf(h2);
    }
  }

  // ---------------- u = h @ I[p] ; h = h - relu(u) ----------------
#pragma unroll
  for (int n = 0; n < NT; ++n) acc2[n] = vzero;
  for (int kc = 0; kc < N; kc += RKC) {
    __syncthreads();
#pragma unroll
    for (int i = 0; i < (RKC * N / 4) / 256; ++i) {
      int idx = tid + i * 256;
      int k = idx / (N / 4), c4 = (idx % (N / 4)) << 2;
      float4 v = *(const float4*)(Ip + (size_t)(kc + k) * N + c4);
      lds_r[(c4 + 0) * LDR + k] = f2bf(v.x);
      lds_r[(c4 + 1) * LDR + k] = f2bf(v.y);
      lds_r[(c4 + 2) * LDR + k] = f2bf(v.z);
      lds_r[(c4 + 3) * LDR + k] = f2bf(v.w);
    }
    __syncthreads();
#pragma unroll
    for (int kb = 0; kb < RKC; kb += KT) {
      v16bf af = load_fragA(lds_h + wave * 16 * LDH + kc + kb, LDH);
      v16bf bfr[NT];
#pragma unroll
      for (int n = 0; n < NT; ++n)
        bfr[n] = load_fragB(lds_r + n * 16 * LDR + kb, LDR);
#pragma unroll
      for (int n = 0; n < NT; ++n) acc2[n] = wmma_bf16(af, bfr[n], acc2[n]);
    }
  }

  // ---------------- gating + store ----------------
#pragma unroll
  for (int n = 0; n < NT; ++n) {
    float av  = Ag[p * N + n * 16 + nl];
    float sg  = 1.f / (1.f + __expf(-av));
#pragma unroll
    for (int r = 0; r < 8; ++r) {
      float u  = acc2[n][r];
      float h3 = acc[n][r] - (u > 0.f ? u : 0.f);
      int row  = wave * 16 + mloc + r;
      out[(size_t)(m0 + row) * (P * N) + p * N + n * 16 + nl] =
          h3 * sg * sel_s[row];
    }
  }
}

// ---------------------------------------------------------------------------
// out[MxN] = A[MxK] @ Bw[KxN] + bias ; N runtime (edge-masked), 64-col tiles
// ---------------------------------------------------------------------------
template <int K>
__global__ __launch_bounds__(256) void gemm_bias_kernel(
    const float* __restrict__ A, const float* __restrict__ Bw,
    const float* __restrict__ bias, float* __restrict__ out, int N) {
  constexpr int MT = 128, NTILE = 64, KT = 32, PAD = 8, LDA = KT + PAD;
  __shared__ __align__(16) __bf16 lds_a[MT * LDA];
  __shared__ __align__(16) __bf16 lds_b[NTILE * LDA];
  const int tid  = threadIdx.x;
  const int wave = tid >> 5;
  const int lane = tid & 31;
  const int nl   = lane & 15;
  const int mloc = (lane >> 4) * 8;
  const int m0   = blockIdx.x * MT;
  const int n0   = blockIdx.y * NTILE;

  const v8f vzero = {0.f, 0.f, 0.f, 0.f, 0.f, 0.f, 0.f, 0.f};
  v8f acc[4];
#pragma unroll
  for (int n = 0; n < 4; ++n) acc[n] = vzero;

  for (int kb = 0; kb < K; kb += KT) {
    __syncthreads();
#pragma unroll
    for (int i = 0; i < (MT * KT / 4) / 256; ++i) {
      int idx = tid + i * 256;
      int m = idx >> 3, k4 = (idx & 7) << 2;
      float4 v = *(const float4*)(A + (size_t)(m0 + m) * K + kb + k4);
      uint2 pk; pk.x = pack_bf2(v.x, v.y); pk.y = pack_bf2(v.z, v.w);
      *(uint2*)(lds_a + m * LDA + k4) = pk;
    }
#pragma unroll
    for (int i = 0; i < (KT * NTILE / 4) / 256; ++i) {
      int idx = tid + i * 256;
      int k = idx >> 4, n4 = (idx & 15) << 2;
      int gn = n0 + n4;
      if (gn + 3 < N) {
        float4 v = *(const float4*)(Bw + (size_t)(kb + k) * N + gn);
        lds_b[(n4 + 0) * LDA + k] = f2bf(v.x);
        lds_b[(n4 + 1) * LDA + k] = f2bf(v.y);
        lds_b[(n4 + 2) * LDA + k] = f2bf(v.z);
        lds_b[(n4 + 3) * LDA + k] = f2bf(v.w);
      } else {
#pragma unroll
        for (int j = 0; j < 4; ++j) {
          int g = gn + j;
          lds_b[(n4 + j) * LDA + k] =
              f2bf(g < N ? Bw[(size_t)(kb + k) * N + g] : 0.f);
        }
      }
    }
    __syncthreads();
    v16bf af = load_fragA(lds_a + wave * 16 * LDA, LDA);
    v16bf bfr[4];
#pragma unroll
    for (int n = 0; n < 4; ++n) bfr[n] = load_fragB(lds_b + n * 16 * LDA, LDA);
#pragma unroll
    for (int n = 0; n < 4; ++n) acc[n] = wmma_bf16(af, bfr[n], acc[n]);
  }
#pragma unroll
  for (int n = 0; n < 4; ++n) {
    int gn = n0 + n * 16 + nl;
    if (gn < N) {
      float bv = bias[gn];
#pragma unroll
      for (int r = 0; r < 8; ++r)
        out[(size_t)(m0 + wave * 16 + mloc + r) * N + gn] = acc[n][r] + bv;
    }
  }
}

// ---------------------------------------------------------------------------
extern "C" void kernel_launch(void* const* d_in, const int* in_sizes, int n_in,
                              void* d_out, int out_size, void* d_ws,
                              size_t ws_size, hipStream_t stream) {
  const float* x     = (const float*)d_in[0];
  const float* W1    = (const float*)d_in[1];
  const float* b1    = (const float*)d_in[2];
  const float* R1    = (const float*)d_in[3];
  const float* I1    = (const float*)d_in[4];
  const float* a1    = (const float*)d_in[5];
  const float* selW1 = (const float*)d_in[6];
  const float* selb1 = (const float*)d_in[7];
  const float* W2    = (const float*)d_in[8];
  const float* b2    = (const float*)d_in[9];
  const float* R2    = (const float*)d_in[10];
  const float* I2    = (const float*)d_in[11];
  const float* a2    = (const float*)d_in[12];
  const float* selW2 = (const float*)d_in[13];
  const float* selb2 = (const float*)d_in[14];
  const float* projW = (const float*)d_in[15];
  const float* projb = (const float*)d_in[16];
  const float* clsW  = (const float*)d_in[17];
  const float* clsb  = (const float*)d_in[18];

  char* ws = (char*)d_ws;
  float* l1   = (float*)(ws + 0);                      // 8192*1024 f32
  float* l2   = (float*)(ws + (size_t)33554432);       // 8192*256  f32
  float* fin  = (float*)(ws + (size_t)41943040);       // 8192*128  f32
  float* sel1 = (float*)(ws + (size_t)46137344);       // 8192*8    f32
  float* sel2 = (float*)(ws + (size_t)46399488);       // 8192*4    f32

  sel_kernel<2048, 8><<<BATCH / 8, 256, 0, stream>>>(x, selW1, selb1, sel1);
  drn_layer_kernel<2048, 8, 128>
      <<<dim3(BATCH / 128, 8), 256, 0, stream>>>(x, W1, b1, R1, I1, a1, sel1, l1);
  sel_kernel<1024, 4><<<BATCH / 8, 256, 0, stream>>>(l1, selW2, selb2, sel2);
  drn_layer_kernel<1024, 4, 64>
      <<<dim3(BATCH / 128, 4), 256, 0, stream>>>(l1, W2, b2, R2, I2, a2, sel2, l2);
  gemm_bias_kernel<256>
      <<<dim3(BATCH / 128, 2), 256, 0, stream>>>(l2, projW, projb, fin, 128);
  gemm_bias_kernel<128>
      <<<dim3(BATCH / 128, 16), 256, 0, stream>>>(fin, clsW, clsb, (float*)d_out, 1000);
}
